// MultiHeadAttention_32238024523798
// MI455X (gfx1250) — matrix operand order repaired and checked by reading, not compiled, not run on hardware
//
#include <hip/hip_runtime.h>
#include <hip/hip_bf16.h>
#include <math.h>
#include <stdint.h>

typedef __bf16 v16bf __attribute__((ext_vector_type(16)));
typedef __bf16 v8bf  __attribute__((ext_vector_type(8)));
typedef float  v8f   __attribute__((ext_vector_type(8)));
typedef int    v4i   __attribute__((ext_vector_type(4)));

#define NB    16
#define CCH   128
#define PIX   4096
#define DDIM  512
#define NHEAD 8
#define LDIM  64

// ---------------------------------------------------------------------------
// gfx1250 async global->LDS path (guarded; falls back to VGPR staging)
// Builtin signature (from hipcc diagnostic): (int4 AS1*, int4 AS3*, imm, imm)
// ---------------------------------------------------------------------------
#if __has_builtin(__builtin_amdgcn_global_load_async_to_lds_b128) && \
    __has_builtin(__builtin_amdgcn_s_wait_asynccnt)
#define USE_ASYNC_LDS 1
#else
#define USE_ASYNC_LDS 0
#endif

#if USE_ASYNC_LDS
typedef v4i __attribute__((address_space(1))) as1_v4i;
typedef v4i __attribute__((address_space(3))) as3_v4i;
// Generic pointer to global memory: bits are the 64-bit VA == AS(1) pointer.
__device__ __forceinline__ as1_v4i* as_glob(const void* p) {
    return (as1_v4i*)(uintptr_t)p;
}
// Generic LDS pointer: low 32 bits are the LDS byte offset (ISA §10.2 aperture
// mapping), which is exactly the AS(3) pointer representation.
__device__ __forceinline__ as3_v4i* as_lds(void* p) {
    return (as3_v4i*)(uint32_t)(uintptr_t)p;
}
#endif

// ---------------------------------------------------------------------------
// f32 -> bf16 elementwise convert (for weight matrices)
// ---------------------------------------------------------------------------
__global__ void cvt_f32_bf16_kernel(const float* __restrict__ src,
                                    __bf16* __restrict__ dst, int n) {
    int i = blockIdx.x * blockDim.x + threadIdx.x;
    if (i < n) dst[i] = (__bf16)src[i];
}

// ---------------------------------------------------------------------------
// WMMA fragment helpers (contiguous k = 16*half + s per lane in A and B, wave32)
// ---------------------------------------------------------------------------
struct FragSet { v16bf a, b0, b1, b2, b3; };

// A fragment from f32 row-major source, converted to bf16 in-register.
__device__ __forceinline__ v16bf loadA_f32(const float* __restrict__ xrow,
                                           int k0, int half) {
    const float* pa0 = xrow + k0 + half * 16;
    const float* pa1 = xrow + k0 + half * 16 + 8;
    v16bf a;
#pragma unroll
    for (int i = 0; i < 8; ++i) {
        a[i]     = (__bf16)pa0[i];
        a[i + 8] = (__bf16)pa1[i];
    }
    return a;
}

// A fragment from bf16 row-major source.
__device__ __forceinline__ v16bf loadA_bf16(const __bf16* __restrict__ arow,
                                            int k0, int half) {
    v8bf lo = *(const v8bf*)(arow + k0 + half * 16);
    v8bf hi = *(const v8bf*)(arow + k0 + half * 16 + 8);
    v16bf a;
#pragma unroll
    for (int i = 0; i < 8; ++i) { a[i] = lo[i]; a[i + 8] = hi[i]; }
    return a;
}

// B fragment: B = W^T with W row-major [N,K] -> contiguous 16 bf16 of W row.
__device__ __forceinline__ v16bf loadB(const __bf16* __restrict__ W,
                                       int n, int k0, int half, int ldk) {
    return *(const v16bf*)(W + (size_t)n * ldk + k0 + half * 16);
}

#define DO_WMMA4(F)                                                                              \
    acc0 = __builtin_amdgcn_wmma_f32_16x16x32_bf16(false, (F).a, false, (F).b0, (short)0, acc0,  \
                                                   false, false);                                \
    acc1 = __builtin_amdgcn_wmma_f32_16x16x32_bf16(false, (F).a, false, (F).b1, (short)0, acc1,  \
                                                   false, false);                                \
    acc2 = __builtin_amdgcn_wmma_f32_16x16x32_bf16(false, (F).a, false, (F).b2, (short)0, acc2,  \
                                                   false, false);                                \
    acc3 = __builtin_amdgcn_wmma_f32_16x16x32_bf16(false, (F).a, false, (F).b3, (short)0, acc3,  \
                                                   false, false)

// ---------------------------------------------------------------------------
// QKV projection GEMM: Y[b] = X[b] (128x4096, f32 row-major) * W^T (W: 512x4096)
// Y: [16,128,512] f32.  bf16 WMMA 16x16x32, f32 accum, double-buffered k-loop.
// grid = (M/16=8, 16 batches), block = 256 (8 waves, each wave 16x64 strip)
// ---------------------------------------------------------------------------
__global__ void __launch_bounds__(256)
gemm_qkv_kernel(const float* __restrict__ X, const __bf16* __restrict__ W,
                float* __restrict__ Y) {
    const int m0   = blockIdx.x * 16;
    const int b    = blockIdx.y;
    const int wave = threadIdx.x >> 5;
    const int lane = threadIdx.x & 31;
    const int lm   = lane & 15;
    const int half = lane >> 4;
    const int n0   = wave * 64;          // 8 waves * 64 = 512 = N

    const float* xrow = X + ((size_t)b * CCH + (m0 + lm)) * PIX;
    v8f acc0 = {}, acc1 = {}, acc2 = {}, acc3 = {};

#define LOAD_FRAGS_QKV(F, K0)                                \
    do {                                                     \
        (F).a  = loadA_f32(xrow, (K0), half);                \
        (F).b0 = loadB(W, n0 +  0 + lm, (K0), half, PIX);    \
        (F).b1 = loadB(W, n0 + 16 + lm, (K0), half, PIX);    \
        (F).b2 = loadB(W, n0 + 32 + lm, (K0), half, PIX);    \
        (F).b3 = loadB(W, n0 + 48 + lm, (K0), half, PIX);    \
    } while (0)

    FragSet f0, f1;
    LOAD_FRAGS_QKV(f0, 0);
    int k0 = 0;
#pragma unroll 1
    for (; k0 < PIX - 64; k0 += 64) {
        LOAD_FRAGS_QKV(f1, k0 + 32);   // prefetch tile i+1
        DO_WMMA4(f0);                  // compute tile i
        LOAD_FRAGS_QKV(f0, k0 + 64);   // prefetch tile i+2
        DO_WMMA4(f1);                  // compute tile i+1
    }
    LOAD_FRAGS_QKV(f1, PIX - 32);
    DO_WMMA4(f0);
    DO_WMMA4(f1);
#undef LOAD_FRAGS_QKV

    float* yb = Y + (size_t)b * CCH * DDIM;
#pragma unroll
    for (int r = 0; r < 8; ++r) {
        const size_t row = (size_t)(m0 + r + 8 * half) * DDIM;
        yb[row + n0 +  0 + lm] = acc0[r];
        yb[row + n0 + 16 + lm] = acc1[r];
        yb[row + n0 + 32 + lm] = acc2[r];
        yb[row + n0 + 48 + lm] = acc3[r];
    }
}

// ---------------------------------------------------------------------------
// Per-(batch,head) channel attention with online softmax.
// Q,K,V: [16,128,512] f32 (head h occupies cols h*64..h*64+63).
// grid = (8 heads, 16 batches), block = 128 (one thread per channel row).
// K/V staged into LDS via gfx1250 async global->LDS DMA when available.
// ---------------------------------------------------------------------------
__global__ void __launch_bounds__(128)
attn_kernel(const float* __restrict__ Q, const float* __restrict__ K,
            const float* __restrict__ V, const float* __restrict__ temp_ptr,
            float* __restrict__ O) {
    __shared__ float Ks[CCH * LDIM];   // 32 KB
    __shared__ float Vs[CCH * LDIM];   // 32 KB

    const int h = blockIdx.x;
    const int b = blockIdx.y;
    const int t = threadIdx.x;
    const size_t base = (size_t)b * CCH * DDIM + (size_t)h * LDIM;

#if USE_ASYNC_LDS
    // 16B-per-lane async copies; 64-float rows are 4-aligned so chunks never
    // cross rows.  Tracked by ASYNCcnt, no VGPR round-trip.
    for (int i = t * 4; i < CCH * LDIM; i += 128 * 4) {
        const int e = i >> 6, d = i & 63;
        __builtin_amdgcn_global_load_async_to_lds_b128(
            as_glob(K + base + (size_t)e * DDIM + d), as_lds(&Ks[i]), 0, 0);
        __builtin_amdgcn_global_load_async_to_lds_b128(
            as_glob(V + base + (size_t)e * DDIM + d), as_lds(&Vs[i]), 0, 0);
    }
    __builtin_amdgcn_s_wait_asynccnt(0);
    __syncthreads();
#else
    for (int i = t; i < CCH * LDIM; i += 128) {
        const int e = i >> 6, d = i & 63;
        Ks[i] = K[base + (size_t)e * DDIM + d];
        Vs[i] = V[base + (size_t)e * DDIM + d];
    }
    __syncthreads();
#endif

    const float inv_t = 1.0f / temp_ptr[0];
    const float* qp = Q + base + (size_t)t * DDIM;

    float q[LDIM];
#pragma unroll
    for (int d = 0; d < LDIM; ++d) q[d] = qp[d];

    float o[LDIM];
#pragma unroll
    for (int d = 0; d < LDIM; ++d) o[d] = 0.0f;

    float m = -INFINITY, l = 0.0f;
    for (int e = 0; e < CCH; ++e) {
        float s = 0.0f;
#pragma unroll
        for (int d = 0; d < LDIM; ++d) s += q[d] * Ks[e * LDIM + d];
        s *= inv_t;
        const float nm   = fmaxf(m, s);
        const float corr = __expf(m - nm);
        const float p    = __expf(s - nm);
        l = l * corr + p;
#pragma unroll
        for (int d = 0; d < LDIM; ++d) o[d] = o[d] * corr + p * Vs[e * LDIM + d];
        m = nm;
    }

    const float invl = 1.0f / l;
    float* op = O + base + (size_t)t * DDIM;
#pragma unroll
    for (int d = 0; d < LDIM; ++d) op[d] = o[d] * invl;
}

// ---------------------------------------------------------------------------
// Swish -> LayerNorm(D=512, ddof=1, /(std+eps)) -> bf16 output for fc GEMM.
// grid = 2048 rows (b*128+c), block = 128 (4 elements/thread).
// ---------------------------------------------------------------------------
__global__ void __launch_bounds__(128)
swish_ln_kernel(const float* __restrict__ O, const float* __restrict__ gamma,
                const float* __restrict__ beta, __bf16* __restrict__ out) {
    __shared__ float s1[128], s2[128];
    const int row = blockIdx.x;
    const int t   = threadIdx.x;
    const float* rp = O + (size_t)row * DDIM;

    float x[4], sum = 0.0f, sq = 0.0f;
#pragma unroll
    for (int i = 0; i < 4; ++i) {
        const float v = rp[t + 128 * i];
        const float sw = v / (1.0f + __expf(-v));   // v*sigmoid(v)
        x[i] = sw;
        sum += sw;
        sq  += sw * sw;
    }
    s1[t] = sum; s2[t] = sq;
    __syncthreads();
    for (int off = 64; off > 0; off >>= 1) {
        if (t < off) { s1[t] += s1[t + off]; s2[t] += s2[t + off]; }
        __syncthreads();
    }
    const float mean = s1[0] * (1.0f / DDIM);
    const float var  = (s2[0] - (float)DDIM * mean * mean) * (1.0f / (DDIM - 1));
    const float inv  = 1.0f / (sqrtf(fmaxf(var, 0.0f)) + 1e-6f);

    __bf16* op = out + (size_t)row * DDIM;
#pragma unroll
    for (int i = 0; i < 4; ++i) {
        const int d = t + 128 * i;
        op[d] = (__bf16)(gamma[d] * (x[i] - mean) * inv + beta[d]);
    }
}

// ---------------------------------------------------------------------------
// fc GEMM + residual: out[b] = resid[b] + LN[b] (128x512 bf16) * Wfc^T
// Wfc bf16: [4096,512] row-major.  out: [16,128,4096] f32 (pre-BN).
// grid = (N/512=8, M/16=8, 16 batches), block = 256 (8 waves x 64 cols).
// ---------------------------------------------------------------------------
__global__ void __launch_bounds__(256)
gemm_fc_kernel(const __bf16* __restrict__ A, const __bf16* __restrict__ Wfc,
               const float* __restrict__ resid, float* __restrict__ out) {
    const int b    = blockIdx.z;
    const int m0   = blockIdx.y * 16;
    const int wave = threadIdx.x >> 5;
    const int lane = threadIdx.x & 31;
    const int lm   = lane & 15;
    const int half = lane >> 4;
    const int n0   = blockIdx.x * 512 + wave * 64;

    const __bf16* arow = A + ((size_t)b * CCH + (m0 + lm)) * DDIM;
    v8f acc0 = {}, acc1 = {}, acc2 = {}, acc3 = {};

#define LOAD_FRAGS_FC(F, K0)                                  \
    do {                                                      \
        (F).a  = loadA_bf16(arow, (K0), half);                \
        (F).b0 = loadB(Wfc, n0 +  0 + lm, (K0), half, DDIM);  \
        (F).b1 = loadB(Wfc, n0 + 16 + lm, (K0), half, DDIM);  \
        (F).b2 = loadB(Wfc, n0 + 32 + lm, (K0), half, DDIM);  \
        (F).b3 = loadB(Wfc, n0 + 48 + lm, (K0), half, DDIM);  \
    } while (0)

    FragSet f0, f1;
    LOAD_FRAGS_FC(f0, 0);
    int k0 = 0;
#pragma unroll 1
    for (; k0 < DDIM - 64; k0 += 64) {
        LOAD_FRAGS_FC(f1, k0 + 32);
        DO_WMMA4(f0);
        LOAD_FRAGS_FC(f0, k0 + 64);
        DO_WMMA4(f1);
    }
    LOAD_FRAGS_FC(f1, DDIM - 32);
    DO_WMMA4(f0);
    DO_WMMA4(f1);
#undef LOAD_FRAGS_FC

    const size_t bo = (size_t)b * CCH * PIX;
#pragma unroll
    for (int r = 0; r < 8; ++r) {
        const size_t row = bo + (size_t)(m0 + r + 8 * half) * PIX;
        out[row + n0 +  0 + lm] = resid[row + n0 +  0 + lm] + acc0[r];
        out[row + n0 + 16 + lm] = resid[row + n0 + 16 + lm] + acc1[r];
        out[row + n0 + 32 + lm] = resid[row + n0 + 32 + lm] + acc2[r];
        out[row + n0 + 48 + lm] = resid[row + n0 + 48 + lm] + acc3[r];
    }
}

// ---------------------------------------------------------------------------
// BatchNorm stats: per channel over (b,h,w) = 65536 elements, biased var.
// grid = 128 channels, block = 256.  stats[2c]=mean, stats[2c+1]=invstd.
// ---------------------------------------------------------------------------
__global__ void __launch_bounds__(256)
bn_stats_kernel(const float* __restrict__ x, float* __restrict__ stats) {
    __shared__ float s1[256], s2[256];
    const int c = blockIdx.x;
    const int t = threadIdx.x;
    float sum = 0.0f, sq = 0.0f;
    for (int i = t; i < NB * PIX; i += 256) {
        const int bb = i >> 12, p = i & (PIX - 1);
        const float v = x[(size_t)bb * CCH * PIX + (size_t)c * PIX + p];
        sum += v; sq += v * v;
    }
    s1[t] = sum; s2[t] = sq;
    __syncthreads();
    for (int off = 128; off > 0; off >>= 1) {
        if (t < off) { s1[t] += s1[t + off]; s2[t] += s2[t + off]; }
        __syncthreads();
    }
    if (t == 0) {
        const float n    = (float)(NB * PIX);
        const float mean = s1[0] / n;
        const float var  = s2[0] / n - mean * mean;
        stats[2 * c]     = mean;
        stats[2 * c + 1] = rsqrtf(fmaxf(var, 0.0f) + 1e-5f);
    }
}

__global__ void __launch_bounds__(256)
bn_apply_kernel(float* __restrict__ x, const float* __restrict__ stats,
                const float* __restrict__ gamma, const float* __restrict__ beta,
                int n) {
    const int i = blockIdx.x * blockDim.x + threadIdx.x;
    if (i >= n) return;
    const int c = (i >> 12) & (CCH - 1);
    x[i] = (x[i] - stats[2 * c]) * stats[2 * c + 1] * gamma[c] + beta[c];
}

// ---------------------------------------------------------------------------
// Host-side orchestration
// ---------------------------------------------------------------------------
extern "C" void kernel_launch(void* const* d_in, const int* in_sizes, int n_in,
                              void* d_out, int out_size, void* d_ws, size_t ws_size,
                              hipStream_t stream) {
    (void)in_sizes; (void)n_in; (void)out_size; (void)ws_size;

    const float* v_in   = (const float*)d_in[0];
    const float* k_in   = (const float*)d_in[1];
    const float* q_in   = (const float*)d_in[2];
    const float* w_qs   = (const float*)d_in[3];
    const float* w_ks   = (const float*)d_in[4];
    const float* w_vs   = (const float*)d_in[5];
    const float* w_fc   = (const float*)d_in[6];
    const float* ln_g   = (const float*)d_in[7];
    const float* ln_b   = (const float*)d_in[8];
    const float* temp   = (const float*)d_in[9];
    const float* bn_g   = (const float*)d_in[10];
    const float* bn_b   = (const float*)d_in[11];
    float* out          = (float*)d_out;

    // workspace partition
    char* ws = (char*)d_ws;
    const size_t W_ELEM = (size_t)DDIM * PIX;          // 2,097,152
    const size_t QKV_EL = (size_t)NB * CCH * DDIM;     // 1,048,576
    __bf16* Wq_bf  = (__bf16*)ws;                 ws += W_ELEM * 2;
    __bf16* Wk_bf  = (__bf16*)ws;                 ws += W_ELEM * 2;
    __bf16* Wv_bf  = (__bf16*)ws;                 ws += W_ELEM * 2;
    __bf16* Wfc_bf = (__bf16*)ws;                 ws += W_ELEM * 2;
    float*  Qf     = (float*)ws;                  ws += QKV_EL * 4;
    float*  Kf     = (float*)ws;                  ws += QKV_EL * 4;
    float*  Vf     = (float*)ws;                  ws += QKV_EL * 4;
    float*  Of     = (float*)ws;                  ws += QKV_EL * 4;
    __bf16* LN_bf  = (__bf16*)ws;                 ws += QKV_EL * 2;
    float*  stats  = (float*)ws;                  ws += 2 * CCH * 4;

    // 1) weights -> bf16
    {
        const int n = (int)W_ELEM, g = (n + 255) / 256;
        cvt_f32_bf16_kernel<<<g, 256, 0, stream>>>(w_qs, Wq_bf, n);
        cvt_f32_bf16_kernel<<<g, 256, 0, stream>>>(w_ks, Wk_bf, n);
        cvt_f32_bf16_kernel<<<g, 256, 0, stream>>>(w_vs, Wv_bf, n);
        cvt_f32_bf16_kernel<<<g, 256, 0, stream>>>(w_fc, Wfc_bf, n);
    }

    // 2) QKV projections (bf16 WMMA, f32 accum, pipelined)
    {
        dim3 grid(CCH / 16, NB);
        gemm_qkv_kernel<<<grid, 256, 0, stream>>>(q_in, Wq_bf, Qf);
        gemm_qkv_kernel<<<grid, 256, 0, stream>>>(k_in, Wk_bf, Kf);
        gemm_qkv_kernel<<<grid, 256, 0, stream>>>(v_in, Wv_bf, Vf);
    }

    // 3) channel attention (online softmax, async LDS staging)
    attn_kernel<<<dim3(NHEAD, NB), 128, 0, stream>>>(Qf, Kf, Vf, temp, Of);

    // 4) swish + layernorm -> bf16
    swish_ln_kernel<<<NB * CCH, 128, 0, stream>>>(Of, ln_g, ln_b, LN_bf);

    // 5) fc GEMM + residual -> out (pre-BN)
    gemm_fc_kernel<<<dim3(PIX / 512, CCH / 16, NB), 256, 0, stream>>>(
        LN_bf, Wfc_bf, v_in, out);

    // 6) batchnorm (training-mode batch stats), in-place on out
    bn_stats_kernel<<<CCH, 256, 0, stream>>>(out, stats);
    {
        const int n = NB * CCH * PIX;
        bn_apply_kernel<<<(n + 255) / 256, 256, 0, stream>>>(out, stats, bn_g, bn_b, n);
    }
}
